// SparsePlasticityRule_32186484916862
// MI455X (gfx1250) — matrix operand order repaired and checked by reading, not compiled, not run on hardware
//
#include <hip/hip_runtime.h>
#include <hip/hip_bf16.h>

// Problem constants (from reference): B=512, P=Q=2048
#define K_B     512
#define P_DIM   2048
#define Q_DIM   2048
#define PQ      (P_DIM * Q_DIM)
#define TARGET_UPDATES 419430u   // int(0.1 * 2048*2048)

// GEMM tiling
#define TILE_M  64
#define TILE_N  128
#define TILE_K  32
#define A_ST    34   // padded LDS row stride (bf16 elems) to spread banks
#define B_ST    34
#define NBINS   4096 // histogram bins on |x| bits >> 19

typedef __attribute__((ext_vector_type(16))) __bf16 v16bf;
typedef __attribute__((ext_vector_type(8)))  float  v8f;

union Frag { unsigned int u[8]; v16bf v; };

__device__ __forceinline__ unsigned short f32_bf16_rne(float x) {
  unsigned int u = __builtin_bit_cast(unsigned int, x);
  u += 0x7fffu + ((u >> 16) & 1u);
  return (unsigned short)(u >> 16);
}
__device__ __forceinline__ float bf16_f32(unsigned short h) {
  unsigned int u = ((unsigned int)h) << 16;
  return __builtin_bit_cast(float, u);
}

// ---------------------------------------------------------------------------
// Kernel 0: zero scratch (hist + counters) — deterministic per launch
// ---------------------------------------------------------------------------
__global__ void zero_ws_kernel(unsigned int* __restrict__ w, int n) {
  int i = blockIdx.x * blockDim.x + threadIdx.x;
  if (i < n) w[i] = 0u;
}

// ---------------------------------------------------------------------------
// Kernel 1: split-bf16 WMMA GEMM (pre^T @ post) fused with eligibility
// update, activity histogram and active-count.
// Grid: (Q/128, P/64), block 256 (8 waves). Each wave: 2x2 16x16 C tiles.
// ---------------------------------------------------------------------------
__global__ __launch_bounds__(256)
void plasticity_gemm_kernel(const float* __restrict__ pre,    // [B,P]
                            const float* __restrict__ post,   // [B,Q]
                            const float* __restrict__ elig,   // [P,Q]
                            const float* __restrict__ a_plus,
                            const float* __restrict__ a_minus,
                            const float* __restrict__ tau,
                            const float* __restrict__ act_thr,
                            float* __restrict__ out_elig,     // [P,Q]
                            unsigned int* __restrict__ g_hist,
                            unsigned int* __restrict__ g_count)
{
  __shared__ unsigned short sAh[TILE_M * A_ST], sAl[TILE_M * A_ST];
  __shared__ unsigned short sBh[TILE_N * B_ST], sBl[TILE_N * B_ST];
  __shared__ unsigned int   sHist[NBINS];
  __shared__ unsigned int   sCount;

  const int t = threadIdx.x;
  for (int i = t; i < NBINS; i += 256) sHist[i] = 0u;
  if (t == 0) sCount = 0u;

  const int mBase = blockIdx.y * TILE_M;
  const int nBase = blockIdx.x * TILE_N;

  const int wave = t >> 5;
  const int lane = t & 31;
  const int hf   = lane >> 4;   // which 16-lane half of the wave
  const int lm   = lane & 15;
  const int wm   = wave >> 2;   // 0..1 -> 32-row strip
  const int wn   = wave & 3;    // 0..3 -> 32-col strip

  const v8f vzero = {0.f, 0.f, 0.f, 0.f, 0.f, 0.f, 0.f, 0.f};
  v8f acc[2][2];
  #pragma unroll
  for (int i = 0; i < 2; ++i)
    #pragma unroll
    for (int j = 0; j < 2; ++j) acc[i][j] = vzero;

  for (int kb = 0; kb < K_B; kb += TILE_K) {
    // --- stage global -> regs (A: 64x32, 8/thr; B: 32x128, 16/thr) ---
    float aval[8];  int am[8],  ak[8];
    #pragma unroll
    for (int i = 0; i < 8; ++i) {
      int idx = t + i * 256;
      ak[i] = idx >> 6; am[i] = idx & 63;
      aval[i] = pre[(kb + ak[i]) * P_DIM + mBase + am[i]];   // A(m,k)=pre[k][m]
    }
    float bval[16]; int bn[16], bk[16];
    #pragma unroll
    for (int i = 0; i < 16; ++i) {
      int idx = t + i * 256;
      bk[i] = idx >> 7; bn[i] = idx & 127;
      bval[i] = post[(kb + bk[i]) * Q_DIM + nBase + bn[i]];  // B(k,n)=post[k][n]
    }
    __syncthreads();  // previous iteration's LDS reads complete
    // --- convert to bf16 hi/lo split, store K-fastest in LDS ---
    #pragma unroll
    for (int i = 0; i < 8; ++i) {
      unsigned short hi = f32_bf16_rne(aval[i]);
      sAh[am[i] * A_ST + ak[i]] = hi;
      sAl[am[i] * A_ST + ak[i]] = f32_bf16_rne(aval[i] - bf16_f32(hi));
    }
    #pragma unroll
    for (int i = 0; i < 16; ++i) {
      unsigned short hi = f32_bf16_rne(bval[i]);
      sBh[bn[i] * B_ST + bk[i]] = hi;
      sBl[bn[i] * B_ST + bk[i]] = f32_bf16_rne(bval[i] - bf16_f32(hi));
    }
    __syncthreads();

    // --- gather WMMA fragments in contiguous k order (k = 16*hf + s), same for A and B ---
    Frag ah[2], al[2], bh[2], bl[2];
    #pragma unroll
    for (int mt = 0; mt < 2; ++mt) {
      const int row = (wm * 32 + mt * 16 + lm) * A_ST;   // A row M = lm
      #pragma unroll
      for (int v = 0; v < 8; ++v) {
        // A 16x32: lanes 0-15 K=0..15 (pair 2v), lanes 16-31 +16
        const int kk = 2 * v + 16 * hf;
        ah[mt].u[v] = *(const unsigned int*)&sAh[row + kk];
        al[mt].u[v] = *(const unsigned int*)&sAl[row + kk];
      }
    }
    #pragma unroll
    for (int nt = 0; nt < 2; ++nt) {
      const int row = (wn * 32 + nt * 16 + lm) * B_ST;   // B col N = lm
      #pragma unroll
      for (int v = 0; v < 8; ++v) {
        // B 32x16: lanes 0-15 rows K=0..15 (pair 2v), lanes 16-31 +16
        const int kk = 2 * v + 16 * hf;
        bh[nt].u[v] = *(const unsigned int*)&sBh[row + kk];
        bl[nt].u[v] = *(const unsigned int*)&sBl[row + kk];
      }
    }
    // --- 3-term split product: hi*hi + hi*lo + lo*hi (f32 accum) ---
    #pragma unroll
    for (int mt = 0; mt < 2; ++mt)
      #pragma unroll
      for (int nt = 0; nt < 2; ++nt) {
        acc[mt][nt] = __builtin_amdgcn_wmma_f32_16x16x32_bf16(
            false, ah[mt].v, false, bh[nt].v, (short)0, acc[mt][nt], false, false);
        acc[mt][nt] = __builtin_amdgcn_wmma_f32_16x16x32_bf16(
            false, ah[mt].v, false, bl[nt].v, (short)0, acc[mt][nt], false, false);
        acc[mt][nt] = __builtin_amdgcn_wmma_f32_16x16x32_bf16(
            false, al[mt].v, false, bh[nt].v, (short)0, acc[mt][nt], false, false);
      }
  }

  // --- fused epilogue: eligibility update + histogram + active count ---
  const float ap = a_plus[0], amn = a_minus[0];
  const float scale = (ap + amn) * (1.0f / (float)K_B);
  const float decay = __expf(-0.1f / tau[0]);
  const float thr   = act_thr[0];
  unsigned int myCount = 0;

  #pragma unroll
  for (int mt = 0; mt < 2; ++mt)
    #pragma unroll
    for (int nt = 0; nt < 2; ++nt)
      #pragma unroll
      for (int r = 0; r < 8; ++r) {
        // C/D layout: VGPR r -> M = r + 8*hf ; N = lm
        const int gm = mBase + wm * 32 + mt * 16 + hf * 8 + r;
        const int gn = nBase + wn * 32 + nt * 16 + lm;
        const int idx = gm * Q_DIM + gn;
        const float e = acc[mt][nt][r] * scale + elig[idx] * decay;
        out_elig[idx] = e;
        const unsigned int ab = __builtin_bit_cast(unsigned int, e) & 0x7fffffffu;
        if (__builtin_bit_cast(float, ab) > thr) myCount++;
        atomicAdd(&sHist[ab >> 19], 1u);
      }

  atomicAdd(&sCount, myCount);
  __syncthreads();
  for (int i = t; i < NBINS; i += 256) {
    const unsigned int h = sHist[i];
    if (h) atomicAdd(&g_hist[i], h);
  }
  if (t == 0) atomicAdd(g_count, sCount);
}

// ---------------------------------------------------------------------------
// Kernel 2: suffix-scan of histogram -> k-th-largest cut (bin lower edge)
// ---------------------------------------------------------------------------
__global__ __launch_bounds__(256)
void select_threshold_kernel(const unsigned int* __restrict__ g_hist,
                             const unsigned int* __restrict__ g_count,
                             unsigned int* __restrict__ g_tau,
                             unsigned int* __restrict__ g_flag)
{
  __shared__ unsigned int csum[256];
  const int t = threadIdx.x;
  unsigned int s = 0;
  #pragma unroll
  for (int i = 0; i < 16; ++i) s += g_hist[t * 16 + i];
  csum[t] = s;
  __syncthreads();
  if (t == 0) {
    unsigned int cum = 0;
    int bin = 0;
    for (int c = 255; c >= 0; --c) {
      if (cum + csum[c] >= TARGET_UPDATES) {
        for (int b = c * 16 + 15; b >= c * 16; --b) {
          cum += g_hist[b];
          if (cum >= TARGET_UPDATES) { bin = b; break; }
        }
        break;
      }
      cum += csum[c];
    }
    *g_tau  = ((unsigned int)bin) << 19;
    *g_flag = (*g_count > TARGET_UPDATES) ? 1u : 0u;
  }
}

// ---------------------------------------------------------------------------
// Kernel 3: weight_updates = select(threshold, topk-cut, clip) — float4 wide
// ---------------------------------------------------------------------------
__global__ __launch_bounds__(256)
void apply_updates_kernel(const float* __restrict__ out_elig,
                          float* __restrict__ out_w,
                          const float* __restrict__ act_thr,
                          const float* __restrict__ max_w,
                          const unsigned int* __restrict__ g_tau,
                          const unsigned int* __restrict__ g_flag)
{
  const int i = blockIdx.x * blockDim.x + threadIdx.x;
  const float thr = act_thr[0];
  const float mw  = max_w[0];
  const unsigned int taub = *g_tau;
  const unsigned int flag = *g_flag;

  const float4 e4 = ((const float4*)out_elig)[i];
  float r[4] = {e4.x, e4.y, e4.z, e4.w};
  float o[4];
  #pragma unroll
  for (int c = 0; c < 4; ++c) {
    const float e = r[c];
    const unsigned int ab = __builtin_bit_cast(unsigned int, e) & 0x7fffffffu;
    const bool keep = (__builtin_bit_cast(float, ab) > thr) &&
                      (!flag || (ab >= taub));
    const float clipped = fminf(fmaxf(e, -mw), mw);
    o[c] = keep ? clipped : 0.0f;
  }
  ((float4*)out_w)[i] = make_float4(o[0], o[1], o[2], o[3]);
}

// ---------------------------------------------------------------------------
extern "C" void kernel_launch(void* const* d_in, const int* in_sizes, int n_in,
                              void* d_out, int out_size, void* d_ws, size_t ws_size,
                              hipStream_t stream) {
  const float* pre     = (const float*)d_in[0];  // [B,P]
  const float* post    = (const float*)d_in[1];  // [B,Q]
  // d_in[2] = weights — only used for shape/zeros in the reference
  const float* elig    = (const float*)d_in[3];  // [P,Q]
  const float* a_plus  = (const float*)d_in[4];
  const float* a_minus = (const float*)d_in[5];
  const float* tau     = (const float*)d_in[6];
  const float* thr     = (const float*)d_in[7];
  const float* maxw    = (const float*)d_in[8];

  float* out_w = (float*)d_out;                  // output 0: weight_updates
  float* out_e = out_w + (size_t)PQ;             // output 1: new_eligibility

  unsigned int* ws      = (unsigned int*)d_ws;
  unsigned int* g_hist  = ws;                    // [NBINS]
  unsigned int* g_count = ws + NBINS;            // [1]
  unsigned int* g_tau   = ws + NBINS + 1;        // [1]
  unsigned int* g_flag  = ws + NBINS + 2;        // [1]

  zero_ws_kernel<<<(NBINS + 3 + 255) / 256, 256, 0, stream>>>(ws, NBINS + 3);

  dim3 grid(Q_DIM / TILE_N, P_DIM / TILE_M);     // (16, 32)
  plasticity_gemm_kernel<<<grid, 256, 0, stream>>>(
      pre, post, elig, a_plus, a_minus, tau, thr, out_e, g_hist, g_count);

  select_threshold_kernel<<<1, 256, 0, stream>>>(g_hist, g_count, g_tau, g_flag);

  apply_updates_kernel<<<PQ / 4 / 256, 256, 0, stream>>>(
      out_e, out_w, thr, maxw, g_tau, g_flag);
}
